// SoftMaskAttention_32787780337796
// MI455X (gfx1250) — matrix operand order repaired and checked by reading, not compiled, not run on hardware
//
#include <hip/hip_runtime.h>

typedef __attribute__((ext_vector_type(16))) _Float16 v16h;
typedef __attribute__((ext_vector_type(8)))  _Float16 v8h;
typedef __attribute__((ext_vector_type(4)))  _Float16 v4h;
typedef __attribute__((ext_vector_type(8)))  float    v8f;

#define WMMA_F16(a, b, c) \
  __builtin_amdgcn_wmma_f32_16x16x32_f16(false, (a), false, (b), (short)0, (c), false, false)

// ---- Operand loaders (contiguous k order, same in A and B) -------------
// A 16x32 f16: lane (m = lane&15, hf = lane>>4); elem i at K = 16*hf + i
//   -> one contiguous 32B chunk (two 16B loads).
static __device__ inline v16h load_a_f16(const _Float16* __restrict__ X, int ldx,
                                         int k0, int lane) {
  const int m = lane & 15, hf = lane >> 4;
  const _Float16* p0 = X + m * ldx + k0 + hf * 16;
  v8h lo = *(const v8h*)p0;
  v8h hi = *(const v8h*)(p0 + 8);
  v16h a;
#pragma unroll
  for (int i = 0; i < 8; ++i) { a[i] = lo[i]; a[8 + i] = hi[i]; }
  return a;
}

// B 32x16 f16: lane (n = lane&15, hf = lane>>4); elem i at K = 16*hf + i,
// column n  -> one contiguous 32B chunk per lane.  B[k][n] = W[n*ldw + k].
static __device__ inline v16h load_b_f16(const _Float16* __restrict__ W, int ldw,
                                         int k0, int lane) {
  const int n = lane & 15, hf = lane >> 4;
  return *(const v16h*)(W + n * ldw + k0 + hf * 16);
}

// Reductions within each 16-lane half-group (rows live per half-group in C-layout)
static __device__ inline float redmax16(float v) {
#pragma unroll
  for (int m = 1; m < 16; m <<= 1) v = fmaxf(v, __shfl_xor(v, m, 32));
  return v;
}
static __device__ inline float redsum16(float v) {
#pragma unroll
  for (int m = 1; m < 16; m <<= 1) v += __shfl_xor(v, m, 32);
  return v;
}

// ---- Kernel 0: one-shot f32 -> f16 conversion (bandwidth-bound) -------------
__global__ __launch_bounds__(256)
void cvt_f32_f16(const float* __restrict__ src, _Float16* __restrict__ dst, int n4) {
  const int i = blockIdx.x * 256 + threadIdx.x;
  if (i < n4) {
    const float4 v = ((const float4*)src)[i];
    v4h o;
    o[0] = (_Float16)v.x; o[1] = (_Float16)v.y;
    o[2] = (_Float16)v.z; o[3] = (_Float16)v.w;
    ((v4h*)dst)[i] = o;
  }
}

// ---- Kernel 1: projection GEMM  out = X @ W^T + bias, f16 in/out ------------
// Xh: [8192, 768] f16, Wh: [768, 768] f16 (row-major [e,k]).
// transposed==0: dst[((b*12+h)*1024 + l)*64 + d]   (Q/K head-split)
// transposed==1: dst[((b*12+h)*64 + d)*1024 + l]   (V transposed)
// Each wave: 32 rows x 64 cols; per k-step: 12 b128-load clause -> 8 WMMAs.
__global__ __launch_bounds__(128)
void proj_gemm(const _Float16* __restrict__ Xh, const _Float16* __restrict__ Wh,
               const float* __restrict__ bias, _Float16* __restrict__ dst,
               float scale, int transposed) {
  const int lane = threadIdx.x & 31;
  const int wv   = threadIdx.x >> 5;
  const int row0 = (blockIdx.x * 4 + wv) * 32;  // 256 row tiles
  const int e0   = blockIdx.y * 64;             // 12 col groups of 64

  v8f acc[2][4] = {};
  for (int k0 = 0; k0 < 768; k0 += 32) {
    v16h a0 = load_a_f16(Xh + (long)row0 * 768, 768, k0, lane);
    v16h a1 = load_a_f16(Xh + (long)(row0 + 16) * 768, 768, k0, lane);
    v16h bm[4];
#pragma unroll
    for (int g = 0; g < 4; ++g)
      bm[g] = load_b_f16(Wh + (long)(e0 + 16 * g) * 768, 768, k0, lane);
#pragma unroll
    for (int g = 0; g < 4; ++g) {
      acc[0][g] = WMMA_F16(a0, bm[g], acc[0][g]);
      acc[1][g] = WMMA_F16(a1, bm[g], acc[1][g]);
    }
  }

  const int n = lane & 15, hf = lane >> 4;
#pragma unroll
  for (int g = 0; g < 4; ++g) {
    const int e = e0 + 16 * g + n;
    const int h = e >> 6, d = e & 63;
    const float bv = bias[e];
#pragma unroll
    for (int t = 0; t < 2; ++t)
#pragma unroll
      for (int r = 0; r < 8; ++r) {
        const int row = row0 + 16 * t + r + 8 * hf;
        const int bb = row >> 10, l = row & 1023;
        const float val = (acc[t][g][r] + bv) * scale;
        long idx = transposed ? (((long)(bb * 12 + h) * 64 + d) * 1024 + l)
                              : (((long)(bb * 12 + h) * 1024 + l) * 64 + d);
        dst[idx] = (_Float16)val;
      }
  }
}

// ---- Kernel 2: flash attention ---------------------------------------------
// Qb/Kb: [B*H, 1024, 64] f16,  Vt: [B*H, 64, 1024] f16,
// kpm: [B,1024] f32, sf: [12] f32, Ob: [B, 1024, 768] f16
__global__ __launch_bounds__(128)
void attn_kernel(const _Float16* __restrict__ Qb, const _Float16* __restrict__ Kb,
                 const _Float16* __restrict__ Vt, const float* __restrict__ kpm,
                 const float* __restrict__ sf, _Float16* __restrict__ Ob) {
  __shared__ _Float16 lds[4][16 * 32];  // wave-private 16x32 P tile
  const int lane = threadIdx.x & 31;
  const int wv   = threadIdx.x >> 5;
  const int bh   = blockIdx.y;          // 96
  const int b = bh / 12, h = bh % 12;
  const int l0 = (blockIdx.x * 4 + wv) * 16;
  const int n = lane & 15, hf = lane >> 4;

  const _Float16* Qh = Qb + ((long)bh * 1024 + l0) * 64;
  const _Float16* Kh = Kb + (long)bh * 1024 * 64;
  const _Float16* Vh = Vt + (long)bh * 64 * 1024;
  const float sfh = sf[h];
  const float* kpmb = kpm + b * 1024;

  // Q A-operands resident in VGPRs for the whole kernel
  v16h aq0 = load_a_f16(Qh, 64, 0, lane);
  v16h aq1 = load_a_f16(Qh, 64, 32, lane);

  v8f oacc[4] = {};
  float mi[8], li[8];
#pragma unroll
  for (int r = 0; r < 8; ++r) { mi[r] = -3.0e38f; li[r] = 0.0f; }

  _Float16* pt = lds[wv];

  for (int s0 = 0; s0 < 1024; s0 += 32) {
    // prefetch next chunk's K rows / V columns (global_prefetch_b8)
    if (s0 + 32 < 1024) {
      __builtin_prefetch(Kh + (long)(s0 + 32) * 64, 0, 1);
      __builtin_prefetch(Vh + s0 + 32, 0, 1);
    }
    // scores: S[16l x 32s] = Q(16x64) . K^T(64x32)
    v16h bk[4];
    bk[0] = load_b_f16(Kh + (long)s0 * 64, 64, 0, lane);
    bk[1] = load_b_f16(Kh + (long)s0 * 64, 64, 32, lane);
    bk[2] = load_b_f16(Kh + (long)(s0 + 16) * 64, 64, 0, lane);
    bk[3] = load_b_f16(Kh + (long)(s0 + 16) * 64, 64, 32, lane);
    v8f sc0 = {}, sc1 = {};
    sc0 = WMMA_F16(aq0, bk[0], sc0);
    sc0 = WMMA_F16(aq1, bk[1], sc0);
    sc1 = WMMA_F16(aq0, bk[2], sc1);
    sc1 = WMMA_F16(aq1, bk[3], sc1);

    // issue V B-panel loads now so they fly during the softmax VALU work
    v16h bvv[4];
#pragma unroll
    for (int g = 0; g < 4; ++g)
      bvv[g] = load_b_f16(Vh + (long)(16 * g) * 1024, 1024, s0, lane);

    const float bias0 = kpmb[s0 + n] * sfh;
    const float bias1 = kpmb[s0 + 16 + n] * sfh;

    float p0[8], p1[8], scl[8];
#pragma unroll
    for (int r = 0; r < 8; ++r) {
      p0[r] = sc0[r] + bias0;
      p1[r] = sc1[r] + bias1;
      float mx = redmax16(fmaxf(p0[r], p1[r]));
      float mn = fmaxf(mi[r], mx);
      scl[r] = __expf(mi[r] - mn);
      mi[r] = mn;
      p0[r] = __expf(p0[r] - mn);
      p1[r] = __expf(p1[r] - mn);
      li[r] = li[r] * scl[r] + redsum16(p0[r] + p1[r]);
    }
#pragma unroll
    for (int g = 0; g < 4; ++g)
#pragma unroll
      for (int r = 0; r < 8; ++r) oacc[g][r] *= scl[r];

    // stage P (C-layout f32 -> row-major f16 in LDS), then reload in A-layout
#pragma unroll
    for (int r = 0; r < 8; ++r) {
      const int row = r + 8 * hf;
      pt[row * 32 + n]      = (_Float16)p0[r];
      pt[row * 32 + 16 + n] = (_Float16)p1[r];
    }
    v16h ap = load_a_f16(pt, 32, 0, lane);
#pragma unroll
    for (int g = 0; g < 4; ++g) oacc[g] = WMMA_F16(ap, bvv[g], oacc[g]);
  }

  float inv[8];
#pragma unroll
  for (int r = 0; r < 8; ++r) inv[r] = 1.0f / li[r];
  _Float16* Orow = Ob + ((long)b * 1024 + l0) * 768 + h * 64;
#pragma unroll
  for (int g = 0; g < 4; ++g)
#pragma unroll
    for (int r = 0; r < 8; ++r)
      Orow[(long)(r + 8 * hf) * 768 + 16 * g + n] = (_Float16)(oacc[g][r] * inv[r]);
}

// ---- Kernel 3: output projection  out = Ob @ Wo^T + bo  (f32 out) ----------
// 32x64 tile per wave, all-f16 operands.
__global__ __launch_bounds__(128)
void outproj_gemm(const _Float16* __restrict__ Xh, const _Float16* __restrict__ Wh,
                  const float* __restrict__ bias, float* __restrict__ out) {
  const int lane = threadIdx.x & 31;
  const int wv   = threadIdx.x >> 5;
  const int row0 = (blockIdx.x * 4 + wv) * 32;
  const int e0   = blockIdx.y * 64;

  v8f acc[2][4] = {};
  for (int k0 = 0; k0 < 768; k0 += 32) {
    v16h a0 = load_a_f16(Xh + (long)row0 * 768, 768, k0, lane);
    v16h a1 = load_a_f16(Xh + (long)(row0 + 16) * 768, 768, k0, lane);
    v16h bm[4];
#pragma unroll
    for (int g = 0; g < 4; ++g)
      bm[g] = load_b_f16(Wh + (long)(e0 + 16 * g) * 768, 768, k0, lane);
#pragma unroll
    for (int g = 0; g < 4; ++g) {
      acc[0][g] = WMMA_F16(a0, bm[g], acc[0][g]);
      acc[1][g] = WMMA_F16(a1, bm[g], acc[1][g]);
    }
  }
  const int n = lane & 15, hf = lane >> 4;
#pragma unroll
  for (int g = 0; g < 4; ++g) {
    const float bv = bias[e0 + 16 * g + n];
#pragma unroll
    for (int t = 0; t < 2; ++t)
#pragma unroll
      for (int r = 0; r < 8; ++r)
        out[(long)(row0 + 16 * t + r + 8 * hf) * 768 + e0 + 16 * g + n] =
            acc[t][g][r] + bv;
  }
}

extern "C" void kernel_launch(void* const* d_in, const int* in_sizes, int n_in,
                              void* d_out, int out_size, void* d_ws, size_t ws_size,
                              hipStream_t stream) {
  (void)in_sizes; (void)n_in; (void)out_size; (void)ws_size;
  const float* q   = (const float*)d_in[0];
  const float* k   = (const float*)d_in[1];
  const float* v   = (const float*)d_in[2];
  const float* kpm = (const float*)d_in[3];
  const float* Wq  = (const float*)d_in[4];
  const float* bq  = (const float*)d_in[5];
  const float* Wk  = (const float*)d_in[6];
  const float* bk  = (const float*)d_in[7];
  const float* Wv  = (const float*)d_in[8];
  const float* bv  = (const float*)d_in[9];
  const float* Wo  = (const float*)d_in[10];
  const float* bo  = (const float*)d_in[11];
  const float* sf  = (const float*)d_in[12];

  char* ws = (char*)d_ws;
  const size_t XB = (size_t)8 * 1024 * 768 * sizeof(_Float16);  // 12.58 MB
  const size_t WB = (size_t)768 * 768 * sizeof(_Float16);       // 1.18 MB
  _Float16* Qb  = (_Float16*)(ws);               // head-split Q  [B*H,1024,64]
  _Float16* Kb  = (_Float16*)(ws + XB);          // head-split K
  _Float16* Vt  = (_Float16*)(ws + 2 * XB);      // V transposed  [B*H,64,1024]
  _Float16* Ob  = (_Float16*)(ws + 3 * XB);      // attn out      [B,1024,768]
  _Float16* Xq  = (_Float16*)(ws + 4 * XB);      // f16 copies of inputs
  _Float16* Xk  = (_Float16*)(ws + 5 * XB);
  _Float16* Xv  = (_Float16*)(ws + 6 * XB);
  _Float16* Wqh = (_Float16*)(ws + 7 * XB);
  _Float16* Wkh = (_Float16*)(ws + 7 * XB + WB);
  _Float16* Wvh = (_Float16*)(ws + 7 * XB + 2 * WB);
  _Float16* Woh = (_Float16*)(ws + 7 * XB + 3 * WB);

  const int nX4 = 8 * 1024 * 768 / 4;  // 1572864
  const int nW4 = 768 * 768 / 4;       // 147456
  dim3 blk256(256), blk128(128);
  dim3 gX((nX4 + 255) / 256), gW((nW4 + 255) / 256);
  hipLaunchKernelGGL(cvt_f32_f16, gX, blk256, 0, stream, q, Xq, nX4);
  hipLaunchKernelGGL(cvt_f32_f16, gX, blk256, 0, stream, k, Xk, nX4);
  hipLaunchKernelGGL(cvt_f32_f16, gX, blk256, 0, stream, v, Xv, nX4);
  hipLaunchKernelGGL(cvt_f32_f16, gW, blk256, 0, stream, Wq, Wqh, nW4);
  hipLaunchKernelGGL(cvt_f32_f16, gW, blk256, 0, stream, Wk, Wkh, nW4);
  hipLaunchKernelGGL(cvt_f32_f16, gW, blk256, 0, stream, Wv, Wvh, nW4);
  hipLaunchKernelGGL(cvt_f32_f16, gW, blk256, 0, stream, Wo, Woh, nW4);

  dim3 gproj(64, 12);   // 256 row tiles / 4 waves, 12 col groups
  dim3 gattn(16, 96);   // (1024/16)/4 L-tiles, B*H
  hipLaunchKernelGGL(proj_gemm, gproj, blk128, 0, stream, Xq, Wqh, bq, Qb, 0.125f, 0);
  hipLaunchKernelGGL(proj_gemm, gproj, blk128, 0, stream, Xk, Wkh, bk, Kb, 1.0f, 0);
  hipLaunchKernelGGL(proj_gemm, gproj, blk128, 0, stream, Xv, Wvh, bv, Vt, 1.0f, 1);
  hipLaunchKernelGGL(attn_kernel, gattn, blk128, 0, stream, Qb, Kb, Vt, kpm, sf, Ob);
  hipLaunchKernelGGL(outproj_gemm, gproj, blk128, 0, stream, Ob, Woh, bo, (float*)d_out);
}
